// EMAVectorQuantizer_19421842112871
// MI455X (gfx1250) — matrix operand order repaired and checked by reading, not compiled, not run on hardware
//
#include <hip/hip_runtime.h>
#include <hip/hip_bf16.h>

// ---------------------------------------------------------------------------
// VQ codebook search for MI455X (gfx1250, wave32, WMMA + async LDS DMA).
//   z:      [16, 256, 32, 32] f32   (16384 rows x 256 ch, ch-stride = 1024)
//   weight: [8192, 256] f32
// Outputs (concat in d_out, all f32):
//   z_out [16384*256] : channel-last transpose of z
//   z_q   [16384*256] : gathered nearest codebook rows
//   idx   [16384]     : argmin indices (stored as float)
// argmin(||z||^2+||w||^2-2 z.w)  ==  argmax(z.w - 0.5*||w||^2)
// ---------------------------------------------------------------------------

typedef __attribute__((ext_vector_type(16))) _Float16 v16h;
typedef __attribute__((ext_vector_type(8)))  _Float16 v8h;
typedef __attribute__((ext_vector_type(8)))  float    v8f;

#define C_DIM   256
#define N_CODES 8192
#define HW      1024
#define NTILES  (N_CODES / 16)   // 512 (sync kernel)
#define NT2     (N_CODES / 32)   // 256 (async kernel)
#define WROW    272              // LDS weight-tile row stride in halves (544B, 32B-aligned)

// ---------------------------------------------------------------------------
// Kernel 0: wn2[n] = sum_k weight[n][k]^2   (one wave per codebook row)
// ---------------------------------------------------------------------------
__global__ __launch_bounds__(256) void vq_wnorm2_kernel(
    const float* __restrict__ w, float* __restrict__ wn2)
{
  const int row  = blockIdx.x * 8 + (threadIdx.x >> 5);
  const int lane = threadIdx.x & 31;
  const float* p = w + (size_t)row * C_DIM;
  float s = 0.0f;
#pragma unroll
  for (int i = 0; i < 8; ++i) {
    float x = p[lane + 32 * i];
    s += x * x;
  }
#pragma unroll
  for (int m = 16; m >= 1; m >>= 1)
    s += __shfl_xor(s, m, 32);
  if (lane == 0) wn2[row] = s;
}

// ---------------------------------------------------------------------------
// Kernel 0b: codebook f32 -> f16 (one-time, enables async LDS DMA of tiles)
// ---------------------------------------------------------------------------
__global__ __launch_bounds__(256) void vq_cvt_f16_kernel(
    const float* __restrict__ w, _Float16* __restrict__ w16)
{
  const size_t i = ((size_t)blockIdx.x * 256 + threadIdx.x) * 8;
  float4 x0 = *(const float4*)(w + i);
  float4 x1 = *(const float4*)(w + i + 4);
  v8h h;
  h[0] = (_Float16)x0.x; h[1] = (_Float16)x0.y;
  h[2] = (_Float16)x0.z; h[3] = (_Float16)x0.w;
  h[4] = (_Float16)x1.x; h[5] = (_Float16)x1.y;
  h[6] = (_Float16)x1.z; h[7] = (_Float16)x1.w;
  *(v8h*)(w16 + i) = h;
}

// ---------------------------------------------------------------------------
// Kernel 1: z_out = transpose(z) : [b][c][hw] -> [b*hw][c]   (LDS 32x32 tile)
// ---------------------------------------------------------------------------
__global__ __launch_bounds__(256) void vq_transpose_kernel(
    const float* __restrict__ z, float* __restrict__ out)
{
  __shared__ float tile[32][33];
  const int b   = blockIdx.z;
  const int hw0 = blockIdx.x * 32;
  const int c0  = blockIdx.y * 32;
  const int tx  = threadIdx.x & 31;
  const int ty  = threadIdx.x >> 5;   // 0..7

  const float* zb = z + (size_t)b * (C_DIM * HW);
#pragma unroll
  for (int i = 0; i < 4; ++i) {
    int c = ty + i * 8;
    tile[c][tx] = zb[(size_t)(c0 + c) * HW + hw0 + tx];   // coalesced along hw
  }
  __syncthreads();
  float* ob = out + (size_t)b * (C_DIM * HW);
#pragma unroll
  for (int i = 0; i < 4; ++i) {
    int hw = ty + i * 8;
    ob[(size_t)(hw0 + hw) * C_DIM + c0 + tx] = tile[tx][hw]; // coalesced along c
  }
}

// ---------------------------------------------------------------------------
// Shared epilogue: half-wave argmax butterfly + index write + z_q gather.
// C layout: VGPR v -> row v (lanes 0-15) / row v+8 (lanes 16-31), n = lane&15.
// ---------------------------------------------------------------------------
__device__ __forceinline__ void vq_epilogue(
    float best[8], int bidx[8], int m0, int hi, int rl, int lane,
    const float* __restrict__ w, float* __restrict__ zq,
    float* __restrict__ idxOut)
{
#pragma unroll
  for (int v = 0; v < 8; ++v) {
    float bv = best[v];
    int   bi = bidx[v];
#pragma unroll
    for (int m = 1; m <= 8; m <<= 1) {
      float ov = __shfl_xor(bv, m, 32);
      int   oi = __shfl_xor(bi, m, 32);
      if (ov > bv || (ov == bv && oi < bi)) { bv = ov; bi = oi; }
    }
    best[v] = bv;
    bidx[v] = bi;
  }

  if (rl == 0) {
#pragma unroll
    for (int v = 0; v < 8; ++v)
      idxOut[m0 + hi * 8 + v] = (float)bidx[v];
  }

#pragma unroll
  for (int r = 0; r < 16; ++r) {
    int src = __shfl(bidx[r & 7], (r >> 3) * 16, 32);   // row r result holder
    const float4* wq = (const float4*)(w + (size_t)src * C_DIM);
    float4*       oq = (float4*)(zq + (size_t)(m0 + r) * C_DIM);
#pragma unroll
    for (int j = 0; j < 2; ++j)
      oq[lane + 32 * j] = wq[lane + 32 * j];
  }
}

// A fragment loader: 16x256 f32 rows (ch-stride HW) -> 8 x v16h.
// Contiguous k order used here (B reads the same): half 'hi' holds K = kb*32 + hi*16 + (0..15).
__device__ __forceinline__ void vq_load_a(
    const float* __restrict__ zp, int hi, v16h a[8])
{
#pragma unroll
  for (int kb = 0; kb < 8; ++kb) {
#pragma unroll
    for (int j = 0; j < 8; ++j) {
      a[kb][j]     = (_Float16)zp[(size_t)(kb * 32 + hi * 16 + j) * HW];
      a[kb][j + 8] = (_Float16)zp[(size_t)(kb * 32 + hi * 16 + 8 + j) * HW];
    }
  }
}

// ---------------------------------------------------------------------------
// Kernel 2a (preferred): async-DMA pipeline.
// Block = 256 thr = 8 waves; wave = 16 rows; block = 128 rows.
// Per iteration: 32 codes; f16 weight tile (32 x 256) DMA'd into padded LDS
// via global_load_async_to_lds_b128; one s_wait_asynccnt + one barrier.
// ---------------------------------------------------------------------------
__global__ __launch_bounds__(256) void vq_search_async_kernel(
    const float* __restrict__ z, const _Float16* __restrict__ w16,
    const float* __restrict__ w, const float* __restrict__ wn2,
    float* __restrict__ zq, float* __restrict__ idxOut)
{
  __shared__ float    s_wn2[N_CODES];        // 32 KB
  __shared__ _Float16 s_w[2][32 * WROW];     // 2 x 17 KB

  const int t    = threadIdx.x;
  const int lane = t & 31;
  const int wave = t >> 5;
  const int hi   = lane >> 4;
  const int rl   = lane & 15;

#pragma unroll
  for (int i = 0; i < 32; ++i)
    s_wn2[i * 256 + t] = wn2[i * 256 + t];

  const int m0 = blockIdx.x * 128 + wave * 16;
  const int bb = m0 >> 10;
  const int hw = (m0 & 1023) + rl;
  v16h a[8];
  vq_load_a(z + (size_t)bb * (C_DIM * HW) + hw, hi, a);

  // Async staging geometry: thread t owns 64B of row (t>>3): 4 x b128 chunks,
  // consecutive in BOTH global and (padded) LDS -> one addr pair + imm offsets.
  const int srow = t >> 3;               // 0..31
  const int scol = (t & 7) * 32;         // halves
  const unsigned lds0 =
      (unsigned)(uintptr_t)(&s_w[0][srow * WROW + scol]);
  const unsigned lds1 =
      (unsigned)(uintptr_t)(&s_w[1][srow * WROW + scol]);
  const _Float16* gbase = w16 + (size_t)srow * C_DIM + scol;

  // prologue: DMA tile 0 -> buffer 0
  asm volatile(
      "global_load_async_to_lds_b128 %0, %1, off\n\t"
      "global_load_async_to_lds_b128 %0, %1, off offset:16\n\t"
      "global_load_async_to_lds_b128 %0, %1, off offset:32\n\t"
      "global_load_async_to_lds_b128 %0, %1, off offset:48"
      :: "v"(lds0), "v"(gbase) : "memory");
  asm volatile("s_wait_asynccnt 0x0" ::: "memory");
  __syncthreads();

  float best[8];
  int   bidx[8];
#pragma unroll
  for (int v = 0; v < 8; ++v) { best[v] = -3.0e38f; bidx[v] = 0; }

  for (int nt = 0; nt < NT2; ++nt) {
    const int n0  = nt * 32;
    const int buf = nt & 1;

    // DMA next tile into the other buffer (overlaps with WMMA below)
    if (nt + 1 < NT2) {
      const _Float16* gp = gbase + (size_t)(nt + 1) * 32 * C_DIM;
      const unsigned  ld = buf ? lds0 : lds1;
      asm volatile(
          "global_load_async_to_lds_b128 %0, %1, off\n\t"
          "global_load_async_to_lds_b128 %0, %1, off offset:16\n\t"
          "global_load_async_to_lds_b128 %0, %1, off offset:32\n\t"
          "global_load_async_to_lds_b128 %0, %1, off offset:48"
          :: "v"(ld), "v"(gp) : "memory");
    }
    if (nt + 2 < NT2)
      __builtin_prefetch(gbase + (size_t)(nt + 2) * 32 * C_DIM, 0, 1);

    // two 16-code accumulator chains, interleaved for XDL pipelining
    const float c1v = -0.5f * s_wn2[n0 + rl];
    const float c2v = -0.5f * s_wn2[n0 + 16 + rl];
    v8f c1 = { c1v, c1v, c1v, c1v, c1v, c1v, c1v, c1v };
    v8f c2 = { c2v, c2v, c2v, c2v, c2v, c2v, c2v, c2v };

    const _Float16* wrowA = &s_w[buf][rl * WROW + hi * 16];
    const _Float16* wrowB = &s_w[buf][(16 + rl) * WROW + hi * 16];
#pragma unroll
    for (int kb = 0; kb < 8; ++kb) {
      v16h b1 = *(const v16h*)(wrowA + kb * 32);
      v16h b2 = *(const v16h*)(wrowB + kb * 32);
      c1 = __builtin_amdgcn_wmma_f32_16x16x32_f16(
              false, a[kb], false, b1, (short)0, c1, false, false);
      c2 = __builtin_amdgcn_wmma_f32_16x16x32_f16(
              false, a[kb], false, b2, (short)0, c2, false, false);
    }

#pragma unroll
    for (int v = 0; v < 8; ++v) {
      if (c1[v] > best[v]) { best[v] = c1[v]; bidx[v] = n0 + rl; }
      if (c2[v] > best[v]) { best[v] = c2[v]; bidx[v] = n0 + 16 + rl; }
    }

    asm volatile("s_wait_asynccnt 0x0" ::: "memory");
    __syncthreads();
  }

  vq_epilogue(best, bidx, m0, hi, rl, lane, w, zq, idxOut);
}

// ---------------------------------------------------------------------------
// Kernel 2b (fallback if scratch too small for f16 codebook): synchronous
// staging (global->regs->cvt->LDS), 16 codes/iteration, double-buffered.
// ---------------------------------------------------------------------------
__global__ __launch_bounds__(256) void vq_search_kernel(
    const float* __restrict__ z, const float* __restrict__ w,
    const float* __restrict__ wn2, float* __restrict__ zq,
    float* __restrict__ idxOut)
{
  __shared__ float    s_wn2[N_CODES];
  __shared__ _Float16 s_w[2][16 * WROW];

  const int t    = threadIdx.x;
  const int lane = t & 31;
  const int wave = t >> 5;
  const int hi   = lane >> 4;
  const int rl   = lane & 15;

#pragma unroll
  for (int i = 0; i < 32; ++i)
    s_wn2[i * 256 + t] = wn2[i * 256 + t];

  const int m0 = blockIdx.x * 128 + wave * 16;
  const int bb = m0 >> 10;
  const int hw = (m0 & 1023) + rl;
  v16h a[8];
  vq_load_a(z + (size_t)bb * (C_DIM * HW) + hw, hi, a);

  const int tr = t >> 4;
  const int tc = (t & 15) * 16;
  float fr[16];

  {
    const float* wp = w + (size_t)tr * C_DIM + tc;
#pragma unroll
    for (int j = 0; j < 16; ++j) fr[j] = wp[j];
    v8h h0, h1;
#pragma unroll
    for (int j = 0; j < 8; ++j) { h0[j] = (_Float16)fr[j]; h1[j] = (_Float16)fr[j + 8]; }
    _Float16* dst = &s_w[0][tr * WROW + tc];
    *(v8h*)(dst)     = h0;
    *(v8h*)(dst + 8) = h1;
  }
  __syncthreads();

  float best[8];
  int   bidx[8];
#pragma unroll
  for (int v = 0; v < 8; ++v) { best[v] = -3.0e38f; bidx[v] = 0; }

  for (int nt = 0; nt < NTILES; ++nt) {
    const int n0  = nt * 16;
    const int buf = nt & 1;

    if (nt + 1 < NTILES) {
      const float* wp = w + (size_t)((nt + 1) * 16 + tr) * C_DIM + tc;
#pragma unroll
      for (int j = 0; j < 16; ++j) fr[j] = wp[j];
    }
    if (nt + 2 < NTILES)
      __builtin_prefetch(w + (size_t)((nt + 2) * 16 + tr) * C_DIM + tc, 0, 1);

    const float c0v = -0.5f * s_wn2[n0 + rl];
    v8f c = { c0v, c0v, c0v, c0v, c0v, c0v, c0v, c0v };

    const _Float16* wrow = &s_w[buf][rl * WROW + hi * 16];
#pragma unroll
    for (int kb = 0; kb < 8; ++kb) {
      v16h b = *(const v16h*)(wrow + kb * 32);
      c = __builtin_amdgcn_wmma_f32_16x16x32_f16(
            false, a[kb], false, b, (short)0, c, false, false);
    }

    const int ncur = n0 + rl;
#pragma unroll
    for (int v = 0; v < 8; ++v) {
      if (c[v] > best[v]) { best[v] = c[v]; bidx[v] = ncur; }
    }

    __syncthreads();
    if (nt + 1 < NTILES) {
      v8h h0, h1;
#pragma unroll
      for (int j = 0; j < 8; ++j) { h0[j] = (_Float16)fr[j]; h1[j] = (_Float16)fr[j + 8]; }
      _Float16* dst = &s_w[buf ^ 1][tr * WROW + tc];
      *(v8h*)(dst)     = h0;
      *(v8h*)(dst + 8) = h1;
    }
    __syncthreads();
  }

  vq_epilogue(best, bidx, m0, hi, rl, lane, w, zq, idxOut);
}

// ---------------------------------------------------------------------------
extern "C" void kernel_launch(void* const* d_in, const int* in_sizes, int n_in,
                              void* d_out, int out_size, void* d_ws, size_t ws_size,
                              hipStream_t stream) {
  const float* z = (const float*)d_in[0];   // 16*256*32*32
  const float* w = (const float*)d_in[1];   // 8192*256

  float* out   = (float*)d_out;
  float* z_out = out;                              // 16384*256
  float* z_q   = out + (size_t)16384 * 256;        // 16384*256
  float* idxf  = out + (size_t)2 * 16384 * 256;    // 16384

  float*    wn2 = (float*)d_ws;                                  // 32 KB
  _Float16* w16 = (_Float16*)((char*)d_ws + 32768);              // 4 MB
  const size_t need_async = 32768 + (size_t)N_CODES * C_DIM * 2;

  // 1) codebook squared norms
  vq_wnorm2_kernel<<<dim3(N_CODES / 8), dim3(256), 0, stream>>>(w, wn2);

  // 2) z_out transpose (independent)
  vq_transpose_kernel<<<dim3(HW / 32, C_DIM / 32, 16), dim3(256), 0, stream>>>(z, z_out);

  if (ws_size >= need_async) {
    // 3a) one-time f32->f16 codebook, then async-DMA WMMA search
    vq_cvt_f16_kernel<<<dim3((N_CODES * C_DIM) / 2048), dim3(256), 0, stream>>>(w, w16);
    vq_search_async_kernel<<<dim3(16384 / 128), dim3(256), 0, stream>>>(
        z, w16, w, wn2, z_q, idxf);
  } else {
    // 3b) synchronous-staging fallback
    vq_search_kernel<<<dim3(16384 / 128), dim3(256), 0, stream>>>(z, w, wn2, z_q, idxf);
  }
}
